// Expert_47270410059993
// MI455X (gfx1250) — matrix operand order repaired and checked by reading, not compiled, not run on hardware
//
#include <hip/hip_runtime.h>
#include <hip/hip_bf16.h>

#define DEVINL __device__ __forceinline__

typedef __attribute__((ext_vector_type(2)))  float  v2f;
typedef __attribute__((ext_vector_type(8)))  float  v8f;
typedef __bf16 bf16;
typedef __attribute__((ext_vector_type(8)))  bf16   v8bf;
typedef __attribute__((ext_vector_type(16))) bf16   v16bf;

static constexpr int Bn = 8, Np = 4096, Kn = 20;
static constexpr int BN = Bn * Np;

// leaky_relu(y, 0.2) == max(y, 0.2*y) exactly (slope > 0):
// compiles to v_dual_mul :: v_dual_max instead of mul+cmp+cndmask.
DEVINL float leaky(float y) { return fmaxf(y, 0.2f * y); }

// ---------------------------------------------------------------------------
// Async global->LDS staging (CDNA5 GLOBAL_LOAD_ASYNC_TO_LDS_B128, ASYNCcnt)
// ---------------------------------------------------------------------------
DEVINL void async_copy_b128(const void* gsrc, void* lds, int bytes, int tid,
                            int nthreads) {
  unsigned int base = (unsigned int)(unsigned long long)lds;  // low32 = LDS addr
  for (int off = tid * 16; off < bytes; off += nthreads * 16) {
    unsigned int l = base + off;
    const char* g = (const char*)gsrc + off;
    asm volatile("global_load_async_to_lds_b128 %0, %1, off"
                 :: "v"(l), "v"(g) : "memory");
  }
}
DEVINL void wait_async0() { asm volatile("s_wait_asynccnt 0x0" ::: "memory"); }

// ---------------------------------------------------------------------------
// Prep kernels: BN folding, weight transpose/pad + bf16 convert, norms
// ---------------------------------------------------------------------------
__global__ void fold_bn(const float* __restrict__ b, const float* __restrict__ g,
                        const float* __restrict__ bt, const float* __restrict__ m,
                        const float* __restrict__ v, float* __restrict__ s,
                        float* __restrict__ t, int C) {
  int i = blockIdx.x * blockDim.x + threadIdx.x;
  if (i < C) {
    float sc = g[i] * rsqrtf(v[i] + 1e-5f);
    s[i] = sc;
    t[i] = bt[i] + (b[i] - m[i]) * sc;
  }
}

__global__ void head_st(const float* __restrict__ bias, float* __restrict__ s,
                        float* __restrict__ t) {
  int i = threadIdx.x;
  if (i < 16) { s[i] = 1.f; t[i] = (i < 3) ? bias[i] : 0.f; }
}

// Wt[o][c] = bf16(W[c][o]) with zero padding to CinPad/CoutPad.
__global__ void wt_bf16_pad(const float* __restrict__ W, bf16* __restrict__ Wt,
                            int Cin, int CinPad, int Cout, int CoutPad) {
  int i = blockIdx.x * blockDim.x + threadIdx.x;
  if (i >= CoutPad * CinPad) return;
  int o = i / CinPad, c = i - o * CinPad;
  float val = (o < Cout && c < Cin) ? W[(size_t)c * Cout + o] : 0.f;
  Wt[(size_t)o * CinPad + c] = (bf16)val;
}

__global__ void pad_pos_sq(const float* __restrict__ pos, float* __restrict__ pos4,
                           float* __restrict__ sq) {
  int i = blockIdx.x * blockDim.x + threadIdx.x;
  if (i >= BN) return;
  float x = pos[(size_t)i * 3 + 0], y = pos[(size_t)i * 3 + 1], z = pos[(size_t)i * 3 + 2];
  pos4[(size_t)i * 4 + 0] = x; pos4[(size_t)i * 4 + 1] = y;
  pos4[(size_t)i * 4 + 2] = z; pos4[(size_t)i * 4 + 3] = 0.f;
  sq[i] = x * x + y * y + z * z;
}

__global__ void sq64(const float* __restrict__ X, float* __restrict__ sq) {
  int i = blockIdx.x * blockDim.x + threadIdx.x;
  if (i >= BN) return;
  float s = 0.f;
#pragma unroll
  for (int c = 0; c < 64; ++c) { float v = X[(size_t)i * 64 + c]; s += v * v; }
  sq[i] = s;
}

// ---------------------------------------------------------------------------
// KNN: one wave handles 16 query rows; Gram tile via V_WMMA_F32_16X16X4_F32
// (f32 so neighbor selection is exact). Guarded sorted-insert top-20.
// ---------------------------------------------------------------------------
template <int C>
__global__ __launch_bounds__(32) void knn_kernel(const float* __restrict__ X,
                                                 const float* __restrict__ sq,
                                                 int* __restrict__ idx) {
  const int b = blockIdx.y;
  const int row0 = blockIdx.x * 16;
  const int lane = threadIdx.x;
  const int mi = lane & 15;
  const int half = lane >> 4;

  __shared__ float tile[16][17];

  v2f afrag[C / 4];
  const float* arow = X + ((size_t)b * Np + row0 + mi) * C + 2 * half;
#pragma unroll
  for (int s = 0; s < C / 4; ++s) afrag[s] = *(const v2f*)(arow + 4 * s);

  float sqi[8];
#pragma unroll
  for (int v = 0; v < 8; ++v) sqi[v] = sq[(size_t)b * Np + row0 + v + 8 * half];

  float bd[Kn];
  int bi[Kn];
#pragma unroll
  for (int t = 0; t < Kn; ++t) { bd[t] = 3.0e38f; bi[t] = 0; }

#pragma unroll 1
  for (int jt = 0; jt < Np / 16; ++jt) {
    const int col0 = jt * 16;
    v8f acc = {};
    const float* brow = X + ((size_t)b * Np + col0 + mi) * C + 2 * half;
#pragma unroll
    for (int s = 0; s < C / 4; ++s) {
      v2f bfr = *(const v2f*)(brow + 4 * s);
      acc = __builtin_amdgcn_wmma_f32_16x16x4_f32(false, afrag[s], false, bfr,
                                                  (short)0, acc, false, false);
    }
    float sqj = sq[(size_t)b * Np + col0 + mi];
#pragma unroll
    for (int v = 0; v < 8; ++v)
      tile[v + 8 * half][mi] = sqi[v] + sqj - 2.0f * acc[v];
    __syncthreads();
    if (lane < 16) {
      for (int jj = 0; jj < 16; ++jj) {
        float d = tile[lane][jj];
        if (d < bd[Kn - 1]) {
          float cd = d; int ci = col0 + jj;
#pragma unroll
          for (int p = 0; p < Kn; ++p) {
            if (cd < bd[p]) {
              float tf = bd[p]; bd[p] = cd; cd = tf;
              int ti = bi[p]; bi[p] = ci; ci = ti;
            }
          }
        }
      }
    }
    __syncthreads();
  }
  if (lane < 16) {
    int* op = idx + ((size_t)b * Np + row0 + lane) * Kn;
#pragma unroll
    for (int t = 0; t < Kn; ++t) op[t] = bi[t];
  }
}

// ---------------------------------------------------------------------------
// bf16 WMMA fragment helper (A 16x32, contiguous k = 16*half + s; B uses the same order).
// ---------------------------------------------------------------------------
DEVINL v16bf a_frag(const bf16* bufI, int Cin, int m, int ks, int half) {
  const bf16* pa = bufI + m * Cin + ks * 32 + 16 * half;
  v8bf lo = *(const v8bf*)pa;
  v8bf hi = *(const v8bf*)(pa + 8);
  return __builtin_shufflevector(lo, hi, 0, 1, 2, 3, 4, 5, 6, 7, 8, 9, 10, 11,
                                 12, 13, 14, 15);
}

// ---------------------------------------------------------------------------
// WMMA edge-conv layers: 32-row tile per point (rows >= Kn are zero pad),
// activations bf16 in LDS, weights bf16 in LDS (async-staged).
// ---------------------------------------------------------------------------
DEVINL void edge_layer(const bf16* bufI, bf16* bufO, const bf16* w,
                       const float* __restrict__ st, int Cin, int Cout,
                       int wave, int nwaves, int lane, int npts) {
  const int half = lane >> 4, nl = lane & 15;
  const int units = npts * (Cout >> 4);
  for (int u = wave; u < units; u += nwaves) {
    int pt = u / (Cout >> 4), nt = u % (Cout >> 4);
    int n = nt * 16 + nl;
    float s = st[n], t = st[Cout + n];
#pragma unroll
    for (int mt = 0; mt < 2; ++mt) {
      v8f acc = {};
      const bf16* arow = bufI + (pt * 32 + mt * 16) * Cin;
      const bf16* wrow = w + (size_t)n * Cin + 16 * half;
      for (int ks = 0; ks < (Cin >> 5); ++ks) {
        v16bf bfr = *(const v16bf*)(wrow + ks * 32);
        v16bf afr = a_frag(arow, Cin, nl, ks, half);
        acc = __builtin_amdgcn_wmma_f32_16x16x32_bf16(false, afr, false, bfr,
                                                      (short)0, acc, false, false);
      }
#pragma unroll
      for (int v = 0; v < 8; ++v) {
        int m = mt * 16 + v + 8 * half;
        bufO[(pt * 32 + m) * Cout + n] = (bf16)leaky(acc[v] * s + t);
      }
    }
  }
}

DEVINL void edge_layer_max(const bf16* bufI, const bf16* w,
                           const float* __restrict__ st, int Cin, int Cout,
                           int wave, int nwaves, int lane, int npts,
                           float* __restrict__ out, int p0, int ostride) {
  const int half = lane >> 4, nl = lane & 15;
  const int units = npts * (Cout >> 4);
  for (int u = wave; u < units; u += nwaves) {
    int pt = u / (Cout >> 4), nt = u % (Cout >> 4);
    int n = nt * 16 + nl;
    float s = st[n], t = st[Cout + n];
    float mx = -3.0e38f;
#pragma unroll
    for (int mt = 0; mt < 2; ++mt) {
      v8f acc = {};
      const bf16* arow = bufI + (pt * 32 + mt * 16) * Cin;
      const bf16* wrow = w + (size_t)n * Cin + 16 * half;
      for (int ks = 0; ks < (Cin >> 5); ++ks) {
        v16bf bfr = *(const v16bf*)(wrow + ks * 32);
        v16bf afr = a_frag(arow, Cin, nl, ks, half);
        acc = __builtin_amdgcn_wmma_f32_16x16x32_bf16(false, afr, false, bfr,
                                                      (short)0, acc, false, false);
      }
#pragma unroll
      for (int v = 0; v < 8; ++v) {
        int m = mt * 16 + v + 8 * half;
        if (m < Kn) mx = fmaxf(mx, leaky(acc[v] * s + t));
      }
    }
    mx = fmaxf(mx, __shfl_xor(mx, 16));  // lanes l / l+16 share column n
    if (lane < 16) out[(size_t)(p0 + pt) * ostride + n] = mx;
  }
}

// ---------------------------------------------------------------------------
// Edge conv 1: e=[xi, xj-xi] padded 6->32, MLP 32->64->64->64, max over 20.
// 2 points per 128-thread block; weights async-staged into LDS.
// ---------------------------------------------------------------------------
__global__ __launch_bounds__(128) void edge_conv1_wmma(
    const float* __restrict__ pos, const int* __restrict__ idx,
    const bf16* __restrict__ W1t, const bf16* __restrict__ W2t,
    const bf16* __restrict__ W3t, const float* __restrict__ st,
    float* __restrict__ x1) {
  __shared__ __align__(32) bf16 eb[2 * 32 * 32];
  __shared__ __align__(32) bf16 hA[2 * 32 * 64];
  __shared__ __align__(32) bf16 hB[2 * 32 * 64];
  __shared__ __align__(32) bf16 w1[64 * 32];
  __shared__ __align__(32) bf16 w2[64 * 64];
  __shared__ __align__(32) bf16 w3[64 * 64];
  __shared__ int nbr[2 * Kn];
  const int tid = threadIdx.x, wave = tid >> 5, lane = tid & 31;
  const int p0 = blockIdx.x * 2;
  const int b = p0 / Np;

  async_copy_b128(W1t, w1, 64 * 32 * 2, tid, 128);
  async_copy_b128(W2t, w2, 64 * 64 * 2, tid, 128);
  async_copy_b128(W3t, w3, 64 * 64 * 2, tid, 128);
  if (tid < 2 * Kn) nbr[tid] = idx[(size_t)p0 * Kn + tid];
  __syncthreads();  // nbr visible

  for (int u = tid; u < 2 * 32 * 32; u += 128) {
    int pt = u >> 10, row = (u >> 5) & 31, c = u & 31;
    float val = 0.f;
    if (row < Kn && c < 6) {
      size_t pi = (size_t)(p0 + pt);
      float xi = pos[pi * 3 + (c < 3 ? c : c - 3)];
      if (c < 3) {
        val = xi;
      } else {
        size_t j = (size_t)b * Np + nbr[pt * Kn + row];
        val = pos[j * 3 + (c - 3)] - xi;
      }
    }
    eb[u] = (bf16)val;
  }
  wait_async0();
  __syncthreads();  // eb + async-staged weights ready

  edge_layer(eb, hA, w1, st, 32, 64, wave, 4, lane, 2);
  __syncthreads();
  edge_layer(hA, hB, w2, st + 128, 64, 64, wave, 4, lane, 2);
  __syncthreads();
  edge_layer_max(hB, w3, st + 256, 64, 64, wave, 4, lane, 2, x1, p0, 64);
}

// ---------------------------------------------------------------------------
// Edge conv 2: e=[xi, xj-xi] (128), layer 128->128 + BN + leaky + max.
// ---------------------------------------------------------------------------
__global__ __launch_bounds__(128) void edge_conv2_wmma(
    const float* __restrict__ x1, const int* __restrict__ idx,
    const bf16* __restrict__ Wt, const float* __restrict__ st,
    float* __restrict__ x2) {
  __shared__ __align__(32) bf16 eb[2 * 32 * 128];
  __shared__ __align__(32) bf16 wl[128 * 128];
  __shared__ int nbr[2 * Kn];
  const int tid = threadIdx.x, wave = tid >> 5, lane = tid & 31;
  const int p0 = blockIdx.x * 2;
  const int b = p0 / Np;

  async_copy_b128(Wt, wl, 128 * 128 * 2, tid, 128);
  if (tid < 2 * Kn) nbr[tid] = idx[(size_t)p0 * Kn + tid];
  __syncthreads();

  for (int u = tid; u < 2 * 32 * 128; u += 128) {
    int pt = u >> 12, row = (u >> 7) & 31, c = u & 127;
    float val = 0.f;
    if (row < Kn) {
      float xi = x1[(size_t)(p0 + pt) * 64 + (c & 63)];
      if (c < 64) {
        val = xi;
      } else {
        size_t j = (size_t)b * Np + nbr[pt * Kn + row];
        val = x1[j * 64 + (c - 64)] - xi;
      }
    }
    eb[u] = (bf16)val;
  }
  wait_async0();
  __syncthreads();

  edge_layer_max(eb, wl, st, 128, 128, wave, 4, lane, 2, x2, p0, 128);
}

// ---------------------------------------------------------------------------
// Fused point MLP 192->1024->512->256->3, 64 points per block (MT=4 M-tiles),
// one B-fragment feeds 4 accumulator chains => 4x arithmetic intensity vs L2.
// LDS: 128KB wide + 64KB narrow activation buffers (CDNA5 allows 320KB/WG).
// ---------------------------------------------------------------------------
DEVINL void mlp4(const bf16* bufI, bf16* bufO, const bf16* __restrict__ Wt,
                 const float* __restrict__ st, int Cin, int Cout, int wave,
                 int nwaves, int lane) {
  const int half = lane >> 4, nl = lane & 15;
  for (int nt = wave; nt < (Cout >> 4); nt += nwaves) {
    const int n = nt * 16 + nl;
    v8f acc[4];
#pragma unroll
    for (int mt = 0; mt < 4; ++mt) acc[mt] = {};
    const bf16* wrow = Wt + (size_t)n * Cin + 16 * half;
    for (int ks = 0; ks < (Cin >> 5); ++ks) {
      v16bf bfr = *(const v16bf*)(wrow + ks * 32);
#pragma unroll
      for (int mt = 0; mt < 4; ++mt) {
        v16bf afr = a_frag(bufI + mt * 16 * Cin, Cin, nl, ks, half);
        acc[mt] = __builtin_amdgcn_wmma_f32_16x16x32_bf16(
            false, afr, false, bfr, (short)0, acc[mt], false, false);
      }
    }
    float s = st[n], t = st[Cout + n];
#pragma unroll
    for (int mt = 0; mt < 4; ++mt) {
#pragma unroll
      for (int v = 0; v < 8; ++v) {
        int m = mt * 16 + v + 8 * half;
        bufO[m * Cout + n] = (bf16)leaky(acc[mt][v] * s + t);
      }
    }
  }
}

__global__ __launch_bounds__(512) void point_mlp(
    const float* __restrict__ x1, const float* __restrict__ x2,
    const bf16* __restrict__ WtLin, const bf16* __restrict__ WtM1,
    const bf16* __restrict__ WtM2, const bf16* __restrict__ WtHd,
    const float* __restrict__ stLin, const float* __restrict__ stM1,
    const float* __restrict__ stM2, const float* __restrict__ stHd,
    float* __restrict__ out) {
  __shared__ __align__(32) bf16 bufW[64 * 1024];  // 128KB: h1 / h3 / head in
  __shared__ __align__(32) bf16 bufN[64 * 512];   // 64KB: input / h2
  const int tid = threadIdx.x;
  const int wave = tid >> 5, lane = tid & 31;
  const int p0 = blockIdx.x * 64;

  for (int u = tid; u < 64 * 192; u += 512) {  // stage concat(x1,x2) as bf16
    int m = u / 192, c = u - m * 192;
    float v = (c < 64) ? x1[(size_t)(p0 + m) * 64 + c]
                       : x2[(size_t)(p0 + m) * 128 + (c - 64)];
    bufN[m * 192 + c] = (bf16)v;
  }
  __syncthreads();
  mlp4(bufN, bufW, WtLin, stLin, 192, 1024, wave, 16, lane);
  __syncthreads();
  mlp4(bufW, bufN, WtM1, stM1, 1024, 512, wave, 16, lane);
  __syncthreads();
  mlp4(bufN, bufW, WtM2, stM2, 512, 256, wave, 16, lane);
  __syncthreads();
  if (wave < 4) {  // head: 256 -> 3 (cols padded to 16), tanh; mt = wave
    const int half = lane >> 4, nl = lane & 15;
    const int mt = wave;
    v8f acc = {};
    const bf16* wrow = WtHd + (size_t)nl * 256 + 16 * half;
    for (int ks = 0; ks < 8; ++ks) {
      v16bf bfr = *(const v16bf*)(wrow + ks * 32);
      v16bf afr = a_frag(bufW + mt * 16 * 256, 256, nl, ks, half);
      acc = __builtin_amdgcn_wmma_f32_16x16x32_bf16(false, afr, false, bfr,
                                                    (short)0, acc, false, false);
    }
    if (nl < 3) {
      float t = stHd[16 + nl];
#pragma unroll
      for (int v = 0; v < 8; ++v) {
        int m = mt * 16 + v + 8 * half;
        out[(size_t)(p0 + m) * 3 + nl] = tanhf(acc[v] + t);
      }
    }
  }
}

// ---------------------------------------------------------------------------
// Launch. d_in order: pos, batch, conv1(W,b,g,bt,m,v)x3, conv2 x1, lin, mlp1,
// mlp2, headW, headb.
// ---------------------------------------------------------------------------
extern "C" void kernel_launch(void* const* d_in, const int* in_sizes, int n_in,
                              void* d_out, int out_size, void* d_ws,
                              size_t ws_size, hipStream_t stream) {
  (void)in_sizes; (void)n_in; (void)out_size; (void)ws_size;
  auto P = [&](int i) { return (const float*)d_in[i]; };
  const float* pos = P(0);

  char* ws = (char*)d_ws;
  size_t off = 0;
  auto carve = [&](size_t bytes) -> void* {
    void* p = ws + off;
    off += (bytes + 255) & ~(size_t)255;
    return p;
  };
  float* sq     = (float*)carve((size_t)BN * sizeof(float));
  float* pos4   = (float*)carve((size_t)BN * 4 * sizeof(float));
  int*   idx    = (int*)carve((size_t)BN * Kn * sizeof(int));
  float* x1     = (float*)carve((size_t)BN * 64 * sizeof(float));
  float* x2     = (float*)carve((size_t)BN * 128 * sizeof(float));
  float* st_c1  = (float*)carve(6 * 64 * sizeof(float));
  float* st_c2  = (float*)carve(2 * 128 * sizeof(float));
  float* st_lin = (float*)carve(2 * 1024 * sizeof(float));
  float* st_m1  = (float*)carve(2 * 512 * sizeof(float));
  float* st_m2  = (float*)carve(2 * 256 * sizeof(float));
  float* st_hd  = (float*)carve(2 * 16 * sizeof(float));
  bf16* wt_c1a  = (bf16*)carve((size_t)64 * 32 * 2);
  bf16* wt_c1b  = (bf16*)carve((size_t)64 * 64 * 2);
  bf16* wt_c1c  = (bf16*)carve((size_t)64 * 64 * 2);
  bf16* wt_c2   = (bf16*)carve((size_t)128 * 128 * 2);
  bf16* wt_lin  = (bf16*)carve((size_t)1024 * 192 * 2);
  bf16* wt_m1   = (bf16*)carve((size_t)512 * 1024 * 2);
  bf16* wt_m2   = (bf16*)carve((size_t)256 * 512 * 2);
  bf16* wt_hd   = (bf16*)carve((size_t)16 * 256 * 2);

  auto fold = [&](int base, float* s, int C) {
    fold_bn<<<(C + 63) / 64, 64, 0, stream>>>(P(base + 1), P(base + 2),
                                              P(base + 3), P(base + 4),
                                              P(base + 5), s, s + C, C);
  };
  fold(2, st_c1 + 0, 64);
  fold(8, st_c1 + 128, 64);
  fold(14, st_c1 + 256, 64);
  fold(20, st_c2, 128);
  fold(26, st_lin, 1024);
  fold(32, st_m1, 512);
  fold(38, st_m2, 256);
  head_st<<<1, 32, 0, stream>>>(P(45), st_hd, st_hd + 16);

  auto cvt = [&](const float* W, bf16* Wt, int Cin, int CinPad, int Cout,
                 int CoutPad) {
    int n = CoutPad * CinPad;
    wt_bf16_pad<<<(n + 255) / 256, 256, 0, stream>>>(W, Wt, Cin, CinPad, Cout,
                                                     CoutPad);
  };
  cvt(P(2), wt_c1a, 6, 32, 64, 64);
  cvt(P(8), wt_c1b, 64, 64, 64, 64);
  cvt(P(14), wt_c1c, 64, 64, 64, 64);
  cvt(P(20), wt_c2, 128, 128, 128, 128);
  cvt(P(26), wt_lin, 192, 192, 1024, 1024);
  cvt(P(32), wt_m1, 1024, 1024, 512, 512);
  cvt(P(38), wt_m2, 512, 512, 256, 256);
  cvt(P(44), wt_hd, 256, 256, 3, 16);

  pad_pos_sq<<<(BN + 255) / 256, 256, 0, stream>>>(pos, pos4, sq);
  knn_kernel<4><<<dim3(Np / 16, Bn), 32, 0, stream>>>(pos4, sq, idx);
  edge_conv1_wmma<<<BN / 2, 128, 0, stream>>>(pos, idx, wt_c1a, wt_c1b, wt_c1c,
                                              st_c1, x1);
  sq64<<<(BN + 255) / 256, 256, 0, stream>>>(x1, sq);
  knn_kernel<64><<<dim3(Np / 16, Bn), 32, 0, stream>>>(x1, sq, idx);
  edge_conv2_wmma<<<BN / 2, 128, 0, stream>>>(x1, idx, wt_c2, st_c2, x2);
  point_mlp<<<BN / 64, 512, 0, stream>>>(x1, x2, wt_lin, wt_m1, wt_m2, wt_hd,
                                         st_lin, st_m1, st_m2, st_hd,
                                         (float*)d_out);
}
